// Mamba2Temporal_64716567216420
// MI455X (gfx1250) — matrix operand order repaired and checked by reading, not compiled, not run on hardware
//
#include <hip/hip_runtime.h>
#include <hip/hip_bf16.h>

// ---------------- types for CDNA5 WMMA ----------------
typedef __attribute__((ext_vector_type(16))) __bf16 v16bf;
typedef __attribute__((ext_vector_type(8)))  float  v8f;
typedef __attribute__((ext_vector_type(2)))  float  v2f;

union FragBF { uint4 u[2]; v16bf v; };

// ---------------- problem constants ----------------
#define TOKENS 4096      // b*l = 2*2048
#define SEQL   2048
#define DM     1024      // d_model
#define DIN    2048      // d_inner
#define NH     32        // heads
#define HD     64        // head dim
#define CD     2176      // conv dim
#define DPJ    4256      // d_in_proj
#define NPAD   4352      // padded to multiple of 256
#define NCHUNK 32
#define CT     64        // chunk length

__device__ __forceinline__ float siluf(float x) { return x / (1.f + __expf(-x)); }

// ======================================================================
// Weight transpose + f32->bf16 conversion (one-time, bandwidth trivial)
// ======================================================================
__global__ void k_wt1(const float* __restrict__ W, __hip_bfloat16* __restrict__ Wt) {
    size_t e = (size_t)blockIdx.x * 256 + threadIdx.x;        // e = n*DM + k
    if (e >= (size_t)NPAD * DM) return;
    int k = (int)(e & (DM - 1));
    int n = (int)(e >> 10);
    float v = (n < DPJ) ? W[(size_t)k * DPJ + n] : 0.f;       // zero-pad cols
    Wt[e] = __float2bfloat16(v);
}

__global__ void k_wt2(const float* __restrict__ W, __hip_bfloat16* __restrict__ Wt) {
    size_t e = (size_t)blockIdx.x * 256 + threadIdx.x;        // e = n*DIN + k
    if (e >= (size_t)DM * DIN) return;
    int k = (int)(e & (DIN - 1));
    int n = (int)(e >> 11);
    Wt[e] = __float2bfloat16(W[(size_t)k * DM + n]);
}

// ======================================================================
// LayerNorm -> bf16 activations (one token per block)
// ======================================================================
__global__ __launch_bounds__(256)
void k_ln(const float* __restrict__ x, const float* __restrict__ w,
          const float* __restrict__ bi, __hip_bfloat16* __restrict__ h) {
    int tok = blockIdx.x, tid = threadIdx.x;
    const float* xr = x + (size_t)tok * DM;
    float v[4], s = 0.f, s2 = 0.f;
    for (int q = 0; q < 4; ++q) {
        v[q] = xr[tid + 256 * q];
        s += v[q]; s2 += v[q] * v[q];
    }
    __shared__ float rs[256], rq[256];
    rs[tid] = s; rq[tid] = s2; __syncthreads();
    for (int o = 128; o > 0; o >>= 1) {
        if (tid < o) { rs[tid] += rs[tid + o]; rq[tid] += rq[tid + o]; }
        __syncthreads();
    }
    float mu  = rs[0] * (1.f / DM);
    float var = rq[0] * (1.f / DM) - mu * mu;
    float inv = rsqrtf(var + 1e-5f);
    for (int q = 0; q < 4; ++q) {
        int i = tid + 256 * q;
        h[(size_t)tok * DM + i] = __float2bfloat16((v[q] - mu) * inv * w[i] + bi[i]);
    }
}

// ======================================================================
// bf16 WMMA GEMM: C[M,N] = A[M,K] * Bt[N,K]^T (+ optional residual)
// block = 256 threads = 8 waves (2 in M x 4 in N)
// block tile 128x256; wave tile 64x64 (4x4 of 16x16) -> 0.5 frag-loads/WMMA
// ======================================================================
__global__ __launch_bounds__(256)
void k_gemm_bf16(const __hip_bfloat16* __restrict__ A,
                 const __hip_bfloat16* __restrict__ Bt,
                 float* __restrict__ C,
                 const float* __restrict__ resid,
                 int K, int ldC) {
    int lane = threadIdx.x & 31;
    int wid  = threadIdx.x >> 5;
    int half = lane >> 4;       // 0 / 1
    int l16  = lane & 15;
    int waveM = wid & 1;        // 0..1
    int waveN = wid >> 1;       // 0..3
    int rowBase = blockIdx.y * 128 + waveM * 64;
    int colBase = blockIdx.x * 256 + waveN * 64;

    v8f z8 = {0, 0, 0, 0, 0, 0, 0, 0};
    v8f acc[4][4];
    for (int mt = 0; mt < 4; ++mt)
        for (int nt = 0; nt < 4; ++nt) acc[mt][nt] = z8;

    for (int k0 = 0; k0 < K; k0 += 32) {
        FragBF a[4];
        // A frag (16x32 bf16): lane<16 -> K=[k0..k0+7]&[k0+16..k0+23]; lane>=16 -> +8
        for (int mt = 0; mt < 4; ++mt) {
            const uint4* pa = (const uint4*)(A + (size_t)(rowBase + mt * 16 + l16) * K
                                               + k0 + half * 8);
            a[mt].u[0] = pa[0];
            a[mt].u[1] = pa[2];     // +16 bf16 elements
        }
        FragBF b[4];
        // B frag (32x16 bf16): lane holds col N=l16, K=[k0..k0+7]&[k0+16..k0+23]; lane>=16 -> +8
        for (int nt = 0; nt < 4; ++nt) {
            const uint4* pb = (const uint4*)(Bt + (size_t)(colBase + nt * 16 + l16) * K
                                               + k0 + half * 8);
            b[nt].u[0] = pb[0];
            b[nt].u[1] = pb[2];
        }
        for (int mt = 0; mt < 4; ++mt)
            for (int nt = 0; nt < 4; ++nt)
                acc[mt][nt] = __builtin_amdgcn_wmma_f32_16x16x32_bf16(
                    false, a[mt].v, false, b[nt].v, (short)0, acc[mt][nt], false, false);
    }
    for (int mt = 0; mt < 4; ++mt)
        for (int nt = 0; nt < 4; ++nt)
            for (int r = 0; r < 8; ++r) {
                int row = rowBase + mt * 16 + r + half * 8;   // C/D layout
                int col = colBase + nt * 16 + l16;
                size_t idx = (size_t)row * ldC + col;
                float v = acc[mt][nt][r];
                if (resid) v += resid[idx];
                C[idx] = v;
            }
}

// ======================================================================
// Depthwise causal conv1d + SiLU over xBC slice of zxbcdt
// ======================================================================
__global__ void k_conv(const float* __restrict__ zx, const float* __restrict__ cw,
                       const float* __restrict__ cb, float* __restrict__ xact) {
    size_t e = (size_t)blockIdx.x * 256 + threadIdx.x;
    if (e >= (size_t)TOKENS * CD) return;
    int ch  = (int)(e % CD);
    int tok = (int)(e / CD);
    int l = tok & (SEQL - 1);
    float acc = cb[ch];
    for (int k = 0; k < 4; ++k) {
        int ll = l - 3 + k;
        if (ll >= 0)
            acc += zx[(size_t)(tok - 3 + k) * NPAD + DIN + ch] * cw[ch * 4 + k];
    }
    xact[e] = siluf(acc);
}

// ======================================================================
// dt = softplus(dt_raw + dt_bias)
// ======================================================================
__global__ void k_dt(const float* __restrict__ zx, const float* __restrict__ dt_bias,
                     float* __restrict__ dtb) {
    int e = blockIdx.x * 256 + threadIdx.x;
    if (e >= TOKENS * NH) return;
    int hh = e & 31, tok = e >> 5;
    float v = zx[(size_t)tok * NPAD + (DIN + CD) + hh] + dt_bias[hh];
    dtb[e] = (v > 20.f) ? v : log1pf(__expf(v));
}

// ======================================================================
// SSD pass1 per (b,c,h): cumsum, G=C*B^T masked by L, Y_diag=(L.G)*X,
// chunk state S = (X.decay)^T * B.  f32 WMMA 16x16x4, operands in LDS.
// 256 threads = 8 waves; each wave owns 2 of 16 16x16 output tiles.
// ======================================================================
__global__ __launch_bounds__(256)
void k_ssd_pass1(const float* __restrict__ xact, const float* __restrict__ dtb,
                 const float* __restrict__ A_log,
                 float* __restrict__ Y, float* __restrict__ Schunk,
                 float* __restrict__ AcsBuf, float* __restrict__ Alast) {
    int c = blockIdx.x, h = blockIdx.y, b = blockIdx.z;
    int tid = threadIdx.x;
    int lane = tid & 31, wid = tid >> 5, half = lane >> 4, l16 = lane & 15;
    int mt = wid & 3;
    int nt0 = (wid >> 2) * 2, nt1 = nt0 + 1;

    __shared__ float sB[64][64];     // Bc (T x n)
    __shared__ float sCG[64][64];    // Cc, later masked G
    __shared__ float sX[64][64];     // X*dt (T x p)
    __shared__ float sAcs[64], sDecay[64], sDt[64];

    float Ah = -__expf(A_log[h]);
    int rowTok0 = b * SEQL + c * CT;

    if (tid < 64) {
        float d = dtb[(size_t)(rowTok0 + tid) * NH + h];
        sDt[tid]  = d;
        sAcs[tid] = d * Ah;
    }
    __syncthreads();
    // inclusive cumsum (Hillis-Steele, 64 elems)
    for (int o = 1; o < 64; o <<= 1) {
        float v = 0.f;
        if (tid < 64 && tid >= o) v = sAcs[tid - o];
        __syncthreads();
        if (tid < 64 && tid >= o) sAcs[tid] += v;
        __syncthreads();
    }
    if (tid < 64) sDecay[tid] = __expf(sAcs[63] - sAcs[tid]);
    if (tid < 64) AcsBuf[(((size_t)b * NCHUNK + c) * NH + h) * 64 + tid] = sAcs[tid];
    if (tid == 0) Alast[((size_t)b * NH + h) * NCHUNK + c] = sAcs[63];
    // stage operands
    for (int idx = tid; idx < 4096; idx += 256) {
        int l = idx >> 6, j = idx & 63;
        size_t rb = (size_t)(rowTok0 + l) * CD;
        sCG[l][j] = xact[rb + (DIN + 64) + j];     // C
        sB[l][j]  = xact[rb + DIN + j];            // B
        sX[l][j]  = xact[rb + h * HD + j] * sDt[l];
    }
    __syncthreads();

    v8f z8 = {0, 0, 0, 0, 0, 0, 0, 0};
    int am  = mt * 16 + l16;
    int bn0 = nt0 * 16 + l16, bn1 = nt1 * 16 + l16;

    // ---- G = C * B^T ----
    v8f g0 = z8, g1 = z8;
    for (int k0 = 0; k0 < 64; k0 += 4) {
        int kb = k0 + half * 2;
        v2f a  = { sCG[am][kb], sCG[am][kb + 1] };
        v2f b0 = { sB[bn0][kb], sB[bn0][kb + 1] };
        v2f b1 = { sB[bn1][kb], sB[bn1][kb + 1] };
        g0 = __builtin_amdgcn_wmma_f32_16x16x4_f32(false, a, false, b0, (short)0, g0, false, false);
        g1 = __builtin_amdgcn_wmma_f32_16x16x4_f32(false, a, false, b1, (short)0, g1, false, false);
    }
    __syncthreads();                  // all C reads done before overwrite
    for (int r = 0; r < 8; ++r) {
        int i = mt * 16 + r + half * 8;
        sCG[i][bn0] = (i >= bn0) ? g0[r] * __expf(sAcs[i] - sAcs[bn0]) : 0.f;
        sCG[i][bn1] = (i >= bn1) ? g1[r] * __expf(sAcs[i] - sAcs[bn1]) : 0.f;
    }
    __syncthreads();

    // ---- Y_diag = (L.G) * X ----
    v8f y0 = z8, y1 = z8;
    for (int k0 = 0; k0 < 64; k0 += 4) {
        int kb = k0 + half * 2;
        v2f a  = { sCG[am][kb], sCG[am][kb + 1] };
        v2f b0 = { sX[kb][bn0], sX[kb + 1][bn0] };
        v2f b1 = { sX[kb][bn1], sX[kb + 1][bn1] };
        y0 = __builtin_amdgcn_wmma_f32_16x16x4_f32(false, a, false, b0, (short)0, y0, false, false);
        y1 = __builtin_amdgcn_wmma_f32_16x16x4_f32(false, a, false, b1, (short)0, y1, false, false);
    }
    for (int r = 0; r < 8; ++r) {
        int i = mt * 16 + r + half * 8;
        Y[(size_t)(rowTok0 + i) * DIN + h * HD + bn0] = y0[r];
        Y[(size_t)(rowTok0 + i) * DIN + h * HD + bn1] = y1[r];
    }

    // ---- S[p][n] = sum_l (X[l][p]*decay[l]) * B[l][n] ----
    v8f s0 = z8, s1 = z8;
    for (int k0 = 0; k0 < 64; k0 += 4) {
        int kb = k0 + half * 2;
        v2f a  = { sX[kb][am] * sDecay[kb], sX[kb + 1][am] * sDecay[kb + 1] };
        v2f b0 = { sB[kb][bn0], sB[kb + 1][bn0] };
        v2f b1 = { sB[kb][bn1], sB[kb + 1][bn1] };
        s0 = __builtin_amdgcn_wmma_f32_16x16x4_f32(false, a, false, b0, (short)0, s0, false, false);
        s1 = __builtin_amdgcn_wmma_f32_16x16x4_f32(false, a, false, b1, (short)0, s1, false, false);
    }
    size_t sb = (((size_t)b * NCHUNK + c) * NH + h) * 4096;
    for (int r = 0; r < 8; ++r) {
        int p = mt * 16 + r + half * 8;
        Schunk[sb + (size_t)p * 64 + bn0] = s0[r];
        Schunk[sb + (size_t)p * 64 + bn1] = s1[r];
    }
}

// ======================================================================
// Inter-chunk scan: init[z] = init[z-1]*exp(Alast[z-1]) + S[z-1]
// one block per (b,h); 16 state entries per thread in registers
// ======================================================================
__global__ __launch_bounds__(256)
void k_chunk_scan(const float* __restrict__ Schunk, const float* __restrict__ Alast,
                  float* __restrict__ Sinit) {
    int bh = blockIdx.x;
    int b = bh >> 5, h = bh & 31;
    int tid = threadIdx.x;
    float st[16];
    for (int q = 0; q < 16; ++q) st[q] = 0.f;
    for (int z = 0; z < NCHUNK; ++z) {
        size_t base = (((size_t)b * NCHUNK + z) * NH + h) * 4096;
        float dec = __expf(Alast[((size_t)b * NH + h) * NCHUNK + z]);
        for (int q = 0; q < 16; ++q) {
            size_t e = base + tid + 256 * q;
            Sinit[e] = st[q];
            st[q] = st[q] * dec + Schunk[e];
        }
    }
}

// ======================================================================
// SSD pass3 per (b,c,h): Y += exp(Acs[l]) * (C * Sinit^T)
// ======================================================================
__global__ __launch_bounds__(256)
void k_ssd_pass3(const float* __restrict__ xact, const float* __restrict__ Sinit,
                 const float* __restrict__ AcsBuf, float* __restrict__ Y) {
    int c = blockIdx.x, h = blockIdx.y, b = blockIdx.z;
    int tid = threadIdx.x;
    int lane = tid & 31, wid = tid >> 5, half = lane >> 4, l16 = lane & 15;
    int mt = wid & 3;
    int nt0 = (wid >> 2) * 2, nt1 = nt0 + 1;

    __shared__ float sC[64][64];
    __shared__ float sS[64][64];
    __shared__ float sE[64];

    int rowTok0 = b * SEQL + c * CT;
    size_t sb = (((size_t)b * NCHUNK + c) * NH + h) * 4096;
    if (tid < 64) sE[tid] = __expf(AcsBuf[(((size_t)b * NCHUNK + c) * NH + h) * 64 + tid]);
    for (int idx = tid; idx < 4096; idx += 256) {
        int l = idx >> 6, j = idx & 63;
        sC[l][j] = xact[(size_t)(rowTok0 + l) * CD + (DIN + 64) + j];
        sS[l][j] = Sinit[sb + idx];       // sS[p][n]
    }
    __syncthreads();

    v8f z8 = {0, 0, 0, 0, 0, 0, 0, 0};
    int am = mt * 16 + l16;
    int bn0 = nt0 * 16 + l16, bn1 = nt1 * 16 + l16;
    v8f o0 = z8, o1 = z8;
    // O[l][p] = sum_n C[l][n] * S[p][n]
    for (int k0 = 0; k0 < 64; k0 += 4) {
        int kb = k0 + half * 2;
        v2f a  = { sC[am][kb], sC[am][kb + 1] };
        v2f b0 = { sS[bn0][kb], sS[bn0][kb + 1] };
        v2f b1 = { sS[bn1][kb], sS[bn1][kb + 1] };
        o0 = __builtin_amdgcn_wmma_f32_16x16x4_f32(false, a, false, b0, (short)0, o0, false, false);
        o1 = __builtin_amdgcn_wmma_f32_16x16x4_f32(false, a, false, b1, (short)0, o1, false, false);
    }
    for (int r = 0; r < 8; ++r) {
        int i = mt * 16 + r + half * 8;
        float e = sE[i];
        Y[(size_t)(rowTok0 + i) * DIN + h * HD + bn0] += o0[r] * e;
        Y[(size_t)(rowTok0 + i) * DIN + h * HD + bn1] += o1[r] * e;
    }
}

// ======================================================================
// D-skip + SiLU gate + gated RMSNorm -> bf16 for GEMM2
// ======================================================================
__global__ __launch_bounds__(256)
void k_gate(const float* __restrict__ Y, const float* __restrict__ xact,
            const float* __restrict__ zx, const float* __restrict__ Dp,
            const float* __restrict__ norm_w, __hip_bfloat16* __restrict__ yg) {
    int tok = blockIdx.x, tid = threadIdx.x;
    float y[8], ss = 0.f;
    for (int q = 0; q < 8; ++q) {
        int i = tid + 256 * q;
        float zv = zx[(size_t)tok * NPAD + i];
        float v = (Y[(size_t)tok * DIN + i] + Dp[i >> 6] * xact[(size_t)tok * CD + i]) * siluf(zv);
        y[q] = v; ss += v * v;
    }
    __shared__ float red[256];
    red[tid] = ss; __syncthreads();
    for (int o = 128; o > 0; o >>= 1) {
        if (tid < o) red[tid] += red[tid + o];
        __syncthreads();
    }
    float scale = rsqrtf(red[0] * (1.f / DIN) + 1e-5f);
    for (int q = 0; q < 8; ++q) {
        int i = tid + 256 * q;
        yg[(size_t)tok * DIN + i] = __float2bfloat16(y[q] * scale * norm_w[i]);
    }
}

// ======================================================================
// launch
// ======================================================================
extern "C" void kernel_launch(void* const* d_in, const int* in_sizes, int n_in,
                              void* d_out, int out_size, void* d_ws, size_t ws_size,
                              hipStream_t stream) {
    (void)in_sizes; (void)n_in; (void)out_size; (void)ws_size;
    const float* x       = (const float*)d_in[0];
    const float* ln_w    = (const float*)d_in[1];
    const float* ln_b    = (const float*)d_in[2];
    const float* Wpin    = (const float*)d_in[3];
    const float* conv_w  = (const float*)d_in[4];
    const float* conv_b  = (const float*)d_in[5];
    const float* dt_bias = (const float*)d_in[6];
    const float* A_log   = (const float*)d_in[7];
    const float* Dp      = (const float*)d_in[8];
    const float* norm_w  = (const float*)d_in[9];
    const float* Wpout   = (const float*)d_in[10];
    float* out = (float*)d_out;

    char* w = (char*)d_ws;
    size_t off = 0;
    auto carve = [&](size_t bytes) -> void* {
        void* p = w + off;
        off += (bytes + 255) & ~(size_t)255;
        return p;
    };
    __hip_bfloat16* hbf  = (__hip_bfloat16*)carve((size_t)TOKENS * DM * 2);
    __hip_bfloat16* Wt1  = (__hip_bfloat16*)carve((size_t)NPAD * DM * 2);
    __hip_bfloat16* Wt2  = (__hip_bfloat16*)carve((size_t)DM * DIN * 2);
    __hip_bfloat16* yg   = (__hip_bfloat16*)carve((size_t)TOKENS * DIN * 2);
    float* zx     = (float*)carve((size_t)TOKENS * NPAD * 4);
    float* xact   = (float*)carve((size_t)TOKENS * CD * 4);
    float* dtb    = (float*)carve((size_t)TOKENS * NH * 4);
    float* Ybuf   = (float*)carve((size_t)TOKENS * DIN * 4);
    float* Schunk = (float*)carve((size_t)2 * NCHUNK * NH * 4096 * 4);
    float* Sinit  = (float*)carve((size_t)2 * NCHUNK * NH * 4096 * 4);
    float* AcsB   = (float*)carve((size_t)2 * NCHUNK * NH * 64 * 4);
    float* Alast  = (float*)carve((size_t)2 * NH * NCHUNK * 4);

    k_wt1<<<(NPAD * DM + 255) / 256, 256, 0, stream>>>(Wpin, Wt1);
    k_wt2<<<(DM * DIN + 255) / 256, 256, 0, stream>>>(Wpout, Wt2);
    k_ln<<<TOKENS, 256, 0, stream>>>(x, ln_w, ln_b, hbf);
    k_gemm_bf16<<<dim3(NPAD / 256, TOKENS / 128), 256, 0, stream>>>(
        hbf, Wt1, zx, nullptr, DM, NPAD);
    k_conv<<<(int)(((size_t)TOKENS * CD + 255) / 256), 256, 0, stream>>>(zx, conv_w, conv_b, xact);
    k_dt<<<(TOKENS * NH + 255) / 256, 256, 0, stream>>>(zx, dt_bias, dtb);
    k_ssd_pass1<<<dim3(NCHUNK, NH, 2), 256, 0, stream>>>(xact, dtb, A_log, Ybuf, Schunk, AcsB, Alast);
    k_chunk_scan<<<2 * NH, 256, 0, stream>>>(Schunk, Alast, Sinit);
    k_ssd_pass3<<<dim3(NCHUNK, NH, 2), 256, 0, stream>>>(xact, Sinit, AcsB, Ybuf);
    k_gate<<<TOKENS, 256, 0, stream>>>(Ybuf, xact, zx, Dp, norm_w, yg);
    k_gemm_bf16<<<dim3(DM / 256, TOKENS / 128), 256, 0, stream>>>(
        yg, Wt2, out, x, DIN, DM);
}
